// TextA3TGCN_27582279975073
// MI455X (gfx1250) — matrix operand order repaired and checked by reading, not compiled, not run on hardware
//
#include <hip/hip_runtime.h>
#include <hip/hip_bf16.h>

// ---------------------------------------------------------------------------
// TextA3TGCN forward for MI455X (gfx1250, wave32).
// All dense math runs through v_wmma_f32_16x16x32_bf16 (bf16 A/B, f32 acc).
#define NN      1024
#define BB      32
#define F_IN    2
#define P_IN    12
#define P_OUT   12
#define HIDD    128
#define EE      16384
#define TEXTD   768
#define ROWS    (BB * NN)          // 32768
#define PER_B   (NN * HIDD)        // 131072 == 2^17, LN reduce size per batch
#define EPSF    1e-12f

typedef __attribute__((ext_vector_type(16))) __bf16    v16bf;
typedef __attribute__((ext_vector_type(8)))  float     v8f;
typedef __attribute__((ext_vector_type(8)))  unsigned  v8u;

// two f32 -> packed bf16x2, round-to-nearest, one v_perm_b32 for the merge
__device__ __forceinline__ unsigned pk2(float f0, float f1) {
  unsigned u0 = __builtin_bit_cast(unsigned, f0) + 0x8000u;
  unsigned u1 = __builtin_bit_cast(unsigned, f1) + 0x8000u;
  // D = { u1[31:16], u0[31:16] } : bytes 2,3 of S1(=u0), bytes 6,7 of {S0,S1}
  return __builtin_amdgcn_perm(u1, u0, 0x07060302u);
}

__device__ __forceinline__ __bf16 f2bf(float f) {
  unsigned u = __builtin_bit_cast(unsigned, f) + 0x8000u;
  unsigned short h = (unsigned short)(u >> 16);
  return __builtin_bit_cast(__bf16, h);
}

__device__ __forceinline__ float gelu_f(float x) {
  return 0.5f * x * (1.0f + erff(x * 0.70710678118654752440f)); // exact erf GELU
}

// ---------------------------------------------------------------------------
// FAST batched WMMA GEMM: requires M % (MT*16)==0, N % (NT*16)==0, K % 64==0
// (K-loop unrolled 2x for WMMA->VALU WAR hazard hiding), lda/ldb/ldc % 4 == 0.
// One wave computes a (MT*16) x (NT*16) strip of C. B fragments reused across
// MT WMMAs, A fragments across NT WMMAs.
template <int MT, int NT, bool TRANSB, bool HASA2, int ACT>
__global__ __launch_bounds__(32) void gemm_fast_kernel(
    const float* __restrict__ A, const float* __restrict__ A2,
    const float* __restrict__ B, const float* __restrict__ bias,
    float* __restrict__ C, int K, int lda, int ldb, int ldc,
    long long sA, long long sB, long long sC, float alpha)
{
  const int tileN = blockIdx.x * (NT << 4);
  const int tileM = blockIdx.y * (MT << 4);
  const long long bat = blockIdx.z;
  A += bat * sA;
  if (HASA2) A2 += bat * sA;
  B += bat * sB;
  C += bat * sC;

  const int lane = threadIdx.x;      // 0..31
  const int half = lane >> 4;        // 0/1
  const int l16  = lane & 15;

  v8f acc[MT][NT] = {};

  // Unroll by 2: two independent fragment register sets in flight fill the
  // 4 co-exec slots of the bf16-WMMA -> VALU WAR hazard with useful work.
#pragma unroll 2
  for (int k0 = 0; k0 < K; k0 += 32) {
    // speculative prefetch of the next A chunk (global_prefetch_b8)
    __builtin_prefetch(A + (long long)(tileM + l16) * lda + k0 + 64, 0, 1);

    // ---- B fragments: element i holds K = k0 + 16*half + i ----
    v16bf bf[NT];
#pragma unroll
    for (int n = 0; n < NT; ++n) {
      const int cn = tileN + (n << 4) + l16;
      v8u bw;
      if (TRANSB) {          // B stored (N x K): consecutive K per lane
        const float* bp = B + (long long)cn * ldb + k0 + (half << 4);
        float4 x0 = *(const float4*)(bp + 0);
        float4 x1 = *(const float4*)(bp + 4);
        float4 x2 = *(const float4*)(bp + 8);
        float4 x3 = *(const float4*)(bp + 12);
        bw[0] = pk2(x0.x, x0.y); bw[1] = pk2(x0.z, x0.w);
        bw[2] = pk2(x1.x, x1.y); bw[3] = pk2(x1.z, x1.w);
        bw[4] = pk2(x2.x, x2.y); bw[5] = pk2(x2.z, x2.w);
        bw[6] = pk2(x3.x, x3.y); bw[7] = pk2(x3.z, x3.w);
      } else {               // B stored (K x N): coalesced-across-lanes rows
        const float* bp = B + (long long)(k0 + (half << 4)) * ldb + cn;
#pragma unroll
        for (int w = 0; w < 8; ++w)
          bw[w] = pk2(bp[(long long)(2 * w) * ldb], bp[(long long)(2 * w + 1) * ldb]);
      }
      bf[n] = __builtin_bit_cast(v16bf, bw);
    }

    // ---- MT A fragments; each fused with NT WMMAs ----
#pragma unroll
    for (int m = 0; m < MT; ++m) {
      const float* ap = A + (long long)(tileM + (m << 4) + l16) * lda + k0 + (half << 4);
      float4 a0 = *(const float4*)(ap + 0);   // K = k0 + 16*half + 0..3
      float4 a1 = *(const float4*)(ap + 4);   // K = k0 + 16*half + 4..7
      float4 a2 = *(const float4*)(ap + 8);   // K = k0 + 16*half + 8..11
      float4 a3 = *(const float4*)(ap + 12);  // K = k0 + 16*half + 12..15
      if (HASA2) {
        const float* qp = A2 + (long long)(tileM + (m << 4) + l16) * lda + k0 + (half << 4);
        float4 q0 = *(const float4*)(qp + 0);
        float4 q1 = *(const float4*)(qp + 4);
        float4 q2 = *(const float4*)(qp + 8);
        float4 q3 = *(const float4*)(qp + 12);
        a0.x += q0.x; a0.y += q0.y; a0.z += q0.z; a0.w += q0.w;
        a1.x += q1.x; a1.y += q1.y; a1.z += q1.z; a1.w += q1.w;
        a2.x += q2.x; a2.y += q2.y; a2.z += q2.z; a2.w += q2.w;
        a3.x += q3.x; a3.y += q3.y; a3.z += q3.z; a3.w += q3.w;
      }
      v8u aw;
      aw[0] = pk2(a0.x, a0.y); aw[1] = pk2(a0.z, a0.w);
      aw[2] = pk2(a1.x, a1.y); aw[3] = pk2(a1.z, a1.w);
      aw[4] = pk2(a2.x, a2.y); aw[5] = pk2(a2.z, a2.w);
      aw[6] = pk2(a3.x, a3.y); aw[7] = pk2(a3.z, a3.w);
      const v16bf af = __builtin_bit_cast(v16bf, aw);
#pragma unroll
      for (int n = 0; n < NT; ++n)
        acc[m][n] = __builtin_amdgcn_wmma_f32_16x16x32_bf16(
            false, af, false, bf[n], (short)0, acc[m][n], false, false);
    }
  }

  // ---- epilogue: alpha, bias, optional GELU; D row = r + 8*half ----
#pragma unroll
  for (int n = 0; n < NT; ++n) {
    const int cn = tileN + (n << 4) + l16;
    const float bsv = bias ? bias[cn] : 0.0f;
#pragma unroll
    for (int m = 0; m < MT; ++m) {
#pragma unroll
      for (int r = 0; r < 8; ++r) {
        int row = tileM + (m << 4) + r + (half << 3);
        float v = acc[m][n][r] * alpha + bsv;
        if (ACT == 1) v = gelu_f(v);
        C[(long long)row * ldc + cn] = v;
      }
    }
  }
}

// ---------------------------------------------------------------------------
// EDGE WMMA GEMM: bounds-checked, scalar loads; only used for the three tiny
// GEMMs with N==12 or K==12 (few hundred MFLOP total). Non-batched, no transB.
template <int ACT>
__global__ __launch_bounds__(32) void gemm_edge_kernel(
    const float* __restrict__ A, const float* __restrict__ B,
    const float* __restrict__ bias, float* __restrict__ C,
    int M, int N, int K, int lda, int ldb, int ldc)
{
  const int tileN = blockIdx.x << 4;
  const int tileM = blockIdx.y << 4;
  const int lane = threadIdx.x;
  const int half = lane >> 4;
  const int l16  = lane & 15;
  const int row = tileM + l16;
  const int col = tileN + l16;

  v8f acc = {};
  for (int k0 = 0; k0 < K; k0 += 32) {
    v16bf a, b;
#pragma unroll
    for (int i = 0; i < 16; ++i) {
      int ka = k0 + (half << 4) + i;
      float av = (row < M && ka < K) ? A[(long long)row * lda + ka] : 0.0f;
      a[i] = f2bf(av);
      int kb = k0 + (half << 4) + i;
      float bv = (col < N && kb < K) ? B[(long long)kb * ldb + col] : 0.0f;
      b[i] = f2bf(bv);
    }
    acc = __builtin_amdgcn_wmma_f32_16x16x32_bf16(
        false, a, false, b, (short)0, acc, false, false);
  }
  float bs = (bias && col < N) ? bias[col] : 0.0f;
#pragma unroll
  for (int r = 0; r < 8; ++r) {
    int m = tileM + r + (half << 3);
    if (m < M && col < N) {
      float v = acc[r] + bs;
      if (ACT == 1) v = gelu_f(v);
      C[(long long)m * ldc + col] = v;
    }
  }
}

// ---------------------------------------------------------------------------
// Small helper kernels (scalar / reduction work)

__global__ void zero_kernel(float* p, long long n) {
  long long i = (long long)blockIdx.x * blockDim.x + threadIdx.x;
  for (; i < n; i += (long long)gridDim.x * blockDim.x) p[i] = 0.0f;
}

__global__ void deg_kernel(const int* __restrict__ dst, const float* __restrict__ w,
                           float* __restrict__ deg) {
  int i = blockIdx.x * blockDim.x + threadIdx.x;
  if (i < EE)            atomicAdd(&deg[dst[i]], w[i]);
  else if (i < EE + NN)  atomicAdd(&deg[i - EE], 1.0f);
}

__global__ void dinv_kernel(const float* __restrict__ deg, float* __restrict__ dinv) {
  int i = blockIdx.x * blockDim.x + threadIdx.x;
  if (i < NN) dinv[i] = (deg[i] > 0.0f) ? rsqrtf(deg[i]) : 0.0f;
}

__global__ void norm_kernel(const int* __restrict__ src, const int* __restrict__ dst,
                            const float* __restrict__ w, const float* __restrict__ dinv,
                            float* __restrict__ norm) {
  int i = blockIdx.x * blockDim.x + threadIdx.x;
  if (i < EE)            norm[i] = dinv[src[i]] * w[i] * dinv[dst[i]];
  else if (i < EE + NN)  { int n = i - EE; norm[i] = dinv[n] * dinv[n]; }
}

// GCN aggregation of the raw F_IN=2 features (linear => aggregate-then-project)
__global__ void scatter2_kernel(const float* __restrict__ x,
                                const int* __restrict__ src, const int* __restrict__ dst,
                                const float* __restrict__ norm, float* __restrict__ agg2) {
  long long gid = (long long)blockIdx.x * blockDim.x + threadIdx.x;
  const long long Etot = EE + NN;
  const long long total = (long long)(P_IN * BB) * Etot;
  if (gid >= total) return;
  int pb = (int)(gid / Etot);
  int e  = (int)(gid % Etot);
  int s, d;
  if (e < EE) { s = src[e]; d = dst[e]; } else { s = d = e - EE; }
  float nw = norm[e];
  int p = pb / BB, b = pb % BB;
  const float* xb = x + (((long long)b * NN + s) * F_IN) * P_IN + p; // x[b,s,f,p]
  float v0 = xb[0];
  float v1 = xb[P_IN];
  float* o = agg2 + ((long long)pb * NN + d) * 2;
  atomicAdd(o + 0, nw * v0);
  atomicAdd(o + 1, nw * v1);
}

// Fold Wc_{z,h} @ Wl_{z,h}[:HID] into 2xHID matrices + bias vectors
__global__ void gate_precomp_kernel(const float* Wcz, const float* bcz,
                                    const float* Wlz, const float* blz,
                                    const float* Wch, const float* bch,
                                    const float* Wlh, const float* blh,
                                    float* Mz, float* vz, float* Mh, float* vh) {
  int j = threadIdx.x;
  if (j >= HIDD) return;
  float m0 = 0, m1 = 0, vv = 0, n0 = 0, n1 = 0, ww = 0;
  for (int i = 0; i < HIDD; ++i) {
    float wl = Wlz[i * HIDD + j];
    m0 += Wcz[i] * wl; m1 += Wcz[HIDD + i] * wl; vv += bcz[i] * wl;
    float wh = Wlh[i * HIDD + j];
    n0 += Wch[i] * wh; n1 += Wch[HIDD + i] * wh; ww += bch[i] * wh;
  }
  Mz[j] = m0; Mz[HIDD + j] = m1; vz[j] = vv + blz[j];
  Mh[j] = n0; Mh[HIDD + j] = n1; vh[j] = ww + blh[j];
}

// h0[b,n,:] = sum_p softmax(attn)_p * (1 - sigmoid(agg2@Mz+vz)) * tanh(agg2@Mh+vh)
__global__ __launch_bounds__(HIDD) void combine_kernel(
    const float* __restrict__ agg2, const float* __restrict__ Mz,
    const float* __restrict__ vz, const float* __restrict__ Mh,
    const float* __restrict__ vh, const float* __restrict__ attn,
    float* __restrict__ h0) {
  int bn = blockIdx.x;
  int b = bn >> 10, n = bn & (NN - 1);
  int j = threadIdx.x;
  float am = attn[0];
  for (int p = 1; p < P_IN; ++p) am = fmaxf(am, attn[p]);
  float wgt[P_IN]; float den = 0.0f;
  for (int p = 0; p < P_IN; ++p) { wgt[p] = expf(attn[p] - am); den += wgt[p]; }
  float mz0 = Mz[j], mz1 = Mz[HIDD + j], vzz = vz[j];
  float mh0 = Mh[j], mh1 = Mh[HIDD + j], vhh = vh[j];
  float acc = 0.0f;
  for (int p = 0; p < P_IN; ++p) {
    const float* a = agg2 + ((long long)(p * BB + b) * NN + n) * 2;
    float a0 = a[0], a1 = a[1];
    float z = 1.0f / (1.0f + expf(-(a0 * mz0 + a1 * mz1 + vzz)));
    float t = tanhf(a0 * mh0 + a1 * mh1 + vhh);
    acc += wgt[p] * (1.0f - z) * t;
  }
  h0[(long long)bn * HIDD + j] = acc / den;
}

// per-batch-sample LayerNorm stats over (N, HID) = 131072 elems
__global__ __launch_bounds__(256) void ln_stats_kernel(const float* __restrict__ X,
                                                       float* __restrict__ stats) {
  __shared__ float ss[256], sq[256];
  int b = blockIdx.x, tid = threadIdx.x;
  const float* p = X + (long long)b * PER_B;
  float s = 0, q = 0;
  for (int i = tid; i < PER_B; i += 256) { float v = p[i]; s += v; q += v * v; }
  ss[tid] = s; sq[tid] = q; __syncthreads();
  for (int st = 128; st > 0; st >>= 1) {
    if (tid < st) { ss[tid] += ss[tid + st]; sq[tid] += sq[tid + st]; }
    __syncthreads();
  }
  if (tid == 0) {
    float mu = ss[0] * (1.0f / PER_B);
    float var = sq[0] * (1.0f / PER_B) - mu * mu;
    stats[2 * b] = mu; stats[2 * b + 1] = var;
  }
}

__global__ void ln_apply_gelu_kernel(const float* __restrict__ X,
                                     const float* __restrict__ g,
                                     const float* __restrict__ bta,
                                     const float* __restrict__ stats,
                                     float* __restrict__ Y) {
  long long i = (long long)blockIdx.x * blockDim.x + threadIdx.x;
  const long long total = (long long)BB * PER_B;
  for (; i < total; i += (long long)gridDim.x * blockDim.x) {
    int b = (int)(i >> 17);          // PER_B == 2^17
    int r = (int)(i & (PER_B - 1));
    float mu = stats[2 * b], var = stats[2 * b + 1];
    float v = (X[i] - mu) * rsqrtf(var + EPSF) * g[r] + bta[r];
    Y[i] = gelu_f(v);
  }
}

__global__ __launch_bounds__(256) void softmax_rows_kernel(float* __restrict__ S, int cols) {
  __shared__ float red[256];
  float* p = S + (long long)blockIdx.x * cols;
  int tid = threadIdx.x;
  float m = -3.4e38f;
  for (int c = tid; c < cols; c += 256) m = fmaxf(m, p[c]);
  red[tid] = m; __syncthreads();
  for (int st = 128; st > 0; st >>= 1) { if (tid < st) red[tid] = fmaxf(red[tid], red[tid + st]); __syncthreads(); }
  m = red[0]; __syncthreads();
  float s = 0.0f;
  for (int c = tid; c < cols; c += 256) { float e = expf(p[c] - m); p[c] = e; s += e; }
  red[tid] = s; __syncthreads();
  for (int st = 128; st > 0; st >>= 1) { if (tid < st) red[tid] += red[tid + st]; __syncthreads(); }
  float inv = 1.0f / red[0];
  for (int c = tid; c < cols; c += 256) p[c] *= inv;
}

// ---------------------------------------------------------------------------
template <int MT, int NT, bool TB, bool A2F, int ACT>
static inline void gemm_fast(hipStream_t st, const float* A, const float* A2,
                             const float* B, const float* bias, float* C,
                             int M, int N, int K, int lda, int ldb, int ldc,
                             long long sA, long long sB, long long sC,
                             int batch, float alpha) {
  dim3 grid(N / (NT * 16), M / (MT * 16), batch);
  gemm_fast_kernel<MT, NT, TB, A2F, ACT><<<grid, dim3(32), 0, st>>>(
      A, A2, B, bias, C, K, lda, ldb, ldc, sA, sB, sC, alpha);
}

template <int ACT>
static inline void gemm_edge(hipStream_t st, const float* A, const float* B,
                             const float* bias, float* C,
                             int M, int N, int K, int lda, int ldb, int ldc) {
  dim3 grid((N + 15) / 16, (M + 15) / 16, 1);
  gemm_edge_kernel<ACT><<<grid, dim3(32), 0, st>>>(A, B, bias, C, M, N, K, lda, ldb, ldc);
}

extern "C" void kernel_launch(void* const* d_in, const int* in_sizes, int n_in,
                              void* d_out, int out_size, void* d_ws, size_t ws_size,
                              hipStream_t stream) {
  // ---- inputs (setup_inputs dict order, params flattened in insertion order)
  const float* x    = (const float*)d_in[0];
  const float* text = (const float*)d_in[1];
  const int*   ei   = (const int*)d_in[2];
  const float* ew   = (const float*)d_in[3];
  const float* attn = (const float*)d_in[4];
  const float* Wc_z = (const float*)d_in[5];
  const float* bc_z = (const float*)d_in[6];
  const float* Wl_z = (const float*)d_in[7];
  const float* bl_z = (const float*)d_in[8];
  const float* Wc_h = (const float*)d_in[9];
  const float* bc_h = (const float*)d_in[10];
  const float* Wl_h = (const float*)d_in[11];
  const float* bl_h = (const float*)d_in[12];
  const float* ln1_g = (const float*)d_in[13];
  const float* ln1_b = (const float*)d_in[14];
  const float* W1 = (const float*)d_in[15];
  const float* b1 = (const float*)d_in[16];
  const float* W2 = (const float*)d_in[17];
  const float* b2 = (const float*)d_in[18];
  const float* Wd1 = (const float*)d_in[19];
  const float* bd1 = (const float*)d_in[20];
  const float* Wd2 = (const float*)d_in[21];
  const float* bd2 = (const float*)d_in[22];
  const float* Wd3 = (const float*)d_in[23];
  const float* bd3 = (const float*)d_in[24];
  const float* Wt1 = (const float*)d_in[25];
  const float* bt1 = (const float*)d_in[26];
  const float* Wt2 = (const float*)d_in[27];
  const float* bt2 = (const float*)d_in[28];
  const float* ln2_g = (const float*)d_in[29];
  const float* ln2_b = (const float*)d_in[30];
  const float* Wk[3] = {(const float*)d_in[31], (const float*)d_in[37], (const float*)d_in[43]};
  const float* bk[3] = {(const float*)d_in[32], (const float*)d_in[38], (const float*)d_in[44]};
  const float* Wv[3] = {(const float*)d_in[33], (const float*)d_in[39], (const float*)d_in[45]};
  const float* bv[3] = {(const float*)d_in[34], (const float*)d_in[40], (const float*)d_in[46]};
  const float* lng[3] = {(const float*)d_in[35], (const float*)d_in[41], (const float*)d_in[47]};
  const float* lnb[3] = {(const float*)d_in[36], (const float*)d_in[42], (const float*)d_in[48]};
  const int* e_src = ei;
  const int* e_dst = ei + EE;
  float* out = (float*)d_out;

  // ---- workspace carve-out (floats; all offsets stay 16B-aligned)
  float* wf = (float*)d_ws;
  size_t off = 0;
  auto alloc = [&](size_t n) { float* p = wf + off; off += n; return p; };
  float* deg   = alloc(NN);
  float* dinv  = alloc(NN);
  float* nrm   = alloc(EE + NN);
  float* Mz = alloc(2 * HIDD); float* vz = alloc(HIDD);
  float* Mh = alloc(2 * HIDD); float* vh = alloc(HIDD);
  float* stats = alloc(64);
  float* agg2  = alloc((size_t)P_IN * BB * NN * 2);
  float* hA    = alloc((size_t)ROWS * HIDD);
  float* hB    = alloc((size_t)ROWS * HIDD);
  float* t1    = alloc((size_t)ROWS * 256);
  float* t2    = alloc((size_t)ROWS * 64);
  float* t3    = alloc((size_t)ROWS * P_OUT);
  float* tq    = alloc((size_t)ROWS * HIDD);
  float* kbuf  = alloc((size_t)ROWS * HIDD);
  float* vbuf  = alloc((size_t)ROWS * HIDD);
  float* attno = alloc((size_t)ROWS * HIDD);
  float* score = alloc((size_t)BB * NN * NN);

  const float scale = 0.0883883476483184f;  // 1/sqrt(128)

  // ---- GCN normalization ----
  zero_kernel<<<64, 256, 0, stream>>>(deg, NN);
  zero_kernel<<<3072, 256, 0, stream>>>(agg2, (long long)P_IN * BB * NN * 2);
  deg_kernel<<<(EE + NN + 255) / 256, 256, 0, stream>>>(e_dst, ew, deg);
  dinv_kernel<<<(NN + 255) / 256, 256, 0, stream>>>(deg, dinv);
  norm_kernel<<<(EE + NN + 255) / 256, 256, 0, stream>>>(e_src, e_dst, ew, dinv, nrm);

  // ---- A3TGCN gates: aggregate F_IN=2 feats, fold Wc@Wl, combine periods ----
  gate_precomp_kernel<<<1, HIDD, 0, stream>>>(Wc_z, bc_z, Wl_z, bl_z,
                                              Wc_h, bc_h, Wl_h, bl_h, Mz, vz, Mh, vh);
  {
    long long tot = (long long)(P_IN * BB) * (EE + NN);
    scatter2_kernel<<<(unsigned)((tot + 255) / 256), 256, 0, stream>>>(x, e_src, e_dst, nrm, agg2);
  }
  combine_kernel<<<ROWS, HIDD, 0, stream>>>(agg2, Mz, vz, Mh, vh, attn, hB);
  ln_stats_kernel<<<BB, 256, 0, stream>>>(hB, stats);
  ln_apply_gelu_kernel<<<4096, 256, 0, stream>>>(hB, ln1_g, ln1_b, stats, hA);
  float* h = hA; float* tmp = hB;

  // ---- text decoder MLP (WMMA fast path) ----
  gemm_fast<4, 2, false, false, 1>(stream, text, nullptr, Wd1, bd1, t1,
                                   ROWS, 256, TEXTD, TEXTD, 256, 256, 0, 0, 0, 1, 1.0f);
  gemm_fast<4, 2, false, false, 1>(stream, t1, nullptr, Wd2, bd2, t2,
                                   ROWS, 64, 256, 256, 64, 64, 0, 0, 0, 1, 1.0f);
  gemm_edge<1>(stream, t2, Wd3, bd3, t3, ROWS, P_OUT, 64, 64, P_OUT, P_OUT);
  gemm_edge<0>(stream, t3, Wt1, bt1, t1, ROWS, HIDD, P_OUT, P_OUT, HIDD, HIDD);
  ln_stats_kernel<<<BB, 256, 0, stream>>>(t1, stats);
  ln_apply_gelu_kernel<<<4096, 256, 0, stream>>>(t1, ln2_g, ln2_b, stats, tq);

  // ---- cross-attention layers over the node axis ----
  for (int i = 0; i < 3; ++i) {
    gemm_fast<4, 2, false, false, 0>(stream, h, nullptr, Wk[i], bk[i], kbuf,
                                     ROWS, HIDD, HIDD, HIDD, HIDD, HIDD, 0, 0, 0, 1, 1.0f);
    gemm_fast<4, 2, false, false, 0>(stream, h, nullptr, Wv[i], bv[i], vbuf,
                                     ROWS, HIDD, HIDD, HIDD, HIDD, HIDD, 0, 0, 0, 1, 1.0f);
    // scores = (tq @ k^T) * scale, batched over B  (transB fast path)
    gemm_fast<4, 2, true, false, 0>(stream, tq, nullptr, kbuf, nullptr, score,
                                    NN, NN, HIDD, HIDD, HIDD, NN,
                                    (long long)NN * HIDD, (long long)NN * HIDD,
                                    (long long)NN * NN, BB, scale);
    softmax_rows_kernel<<<BB * NN, 256, 0, stream>>>(score, NN);
    // attno = scores @ v
    gemm_fast<4, 2, false, false, 0>(stream, score, nullptr, vbuf, nullptr, attno,
                                     NN, HIDD, NN, NN, HIDD, HIDD,
                                     (long long)NN * NN, (long long)NN * HIDD,
                                     (long long)NN * HIDD, BB, 1.0f);
    // u = (h + attn) @ Wv + bv ; h = gelu(ln(u))
    gemm_fast<4, 2, false, true, 0>(stream, h, attno, Wv[i], bv[i], tmp,
                                    ROWS, HIDD, HIDD, HIDD, HIDD, HIDD, 0, 0, 0, 1, 1.0f);
    ln_stats_kernel<<<BB, 256, 0, stream>>>(tmp, stats);
    ln_apply_gelu_kernel<<<4096, 256, 0, stream>>>(tmp, lng[i], lnb[i], stats, h);
  }

  // ---- head ----
  gemm_fast<4, 2, false, false, 1>(stream, h, nullptr, Wt2, bt2, tmp,
                                   ROWS, HIDD, HIDD, HIDD, HIDD, HIDD, 0, 0, 0, 1, 1.0f);
  gemm_fast<4, 2, false, false, 1>(stream, tmp, nullptr, W1, b1, h,
                                   ROWS, HIDD, HIDD, HIDD, HIDD, HIDD, 0, 0, 0, 1, 1.0f);
  gemm_edge<0>(stream, h, W2, b2, out, ROWS, P_OUT, HIDD, HIDD, P_OUT, P_OUT);
}
